// PCELayer_51539607552703
// MI455X (gfx1250) — matrix operand order repaired and checked by reading, not compiled, not run on hardware
//
#include <hip/hip_runtime.h>

// ---------------- problem constants ----------------
constexpr int Bb   = 8;
constexpr int CIN  = 64;
constexpr int COUT = 128;
constexpr int Hh   = 128;
constexpr int Ww   = 128;
constexpr int HW   = Hh * Ww;        // 16384
constexpr int Ee   = 8;
constexpr int HP   = 8;
constexpr int WP   = 8;
constexpr int Gg   = 8;
constexpr float EPSF = 1e-5f;

// ---------------- vector types ----------------
typedef __attribute__((ext_vector_type(16))) __bf16        bf16x16;
typedef __attribute__((ext_vector_type(8)))  float         f32x8;
typedef __attribute__((ext_vector_type(4)))  float         f32x4;
typedef __attribute__((ext_vector_type(4)))  unsigned int  u32x4;
typedef __attribute__((ext_vector_type(2)))  unsigned int  u32x2;
typedef __attribute__((ext_vector_type(8)))  int           i32x8;
typedef __attribute__((ext_vector_type(4)))  int           i32x4;

union FragBF { bf16x16 v; u32x4 q[2]; };

// ---------------- fp32 -> bf16 convert (x and weights) ----------------
__global__ __launch_bounds__(256)
void cvt_bf16_kernel(const float* __restrict__ in, __bf16* __restrict__ out, int n4) {
  int i = blockIdx.x * blockDim.x + threadIdx.x;
  if (i < n4) {
    f32x4 v = ((const f32x4*)in)[i];
    alignas(8) __bf16 o[4];
    o[0] = (__bf16)v[0]; o[1] = (__bf16)v[1];
    o[2] = (__bf16)v[2]; o[3] = (__bf16)v[3];
    ((u32x2*)out)[i] = *(const u32x2*)o;
  }
}

// ---------------- router ----------------
__global__ __launch_bounds__(256)
void router_kernel(const float* __restrict__ x,
                   const float* __restrict__ Rw1, const float* __restrict__ Rb1,
                   const float* __restrict__ Rw2, const float* __restrict__ Rb2,
                   float* __restrict__ gates) {
  __shared__ float part[256];
  __shared__ float gfeat[80];
  __shared__ float hid[128];
  __shared__ float logit[Ee];
  const int t = threadIdx.x;
  const int bid = blockIdx.x;               // b*64 + hp*8 + wp
  const int b = bid >> 6, hp = (bid >> 3) & 7, wp = bid & 7;
  {
    const int c = t & 63, seg = t >> 6;
    const float* xp = x + ((size_t)b * CIN + c) * HW;
    const int r0 = hp * 16 + seg * 4, c0 = wp * 16;
    float s = 0.f;
    for (int r = 0; r < 4; ++r)
      for (int q = 0; q < 16; ++q)
        s += xp[(r0 + r) * Ww + c0 + q];
    part[t] = s;
  }
  __syncthreads();
  if (t < 64) gfeat[t] = (part[t] + part[t + 64] + part[t + 128] + part[t + 192]) * (1.f / 256.f);
  if (t < 16) {   // [sin(ay)x4, cos(ay)x4, sin(ax)x4, cos(ax)x4]
    const int f = t & 3, which = t >> 2;
    const float freq = 3.14159265358979323846f * (float)(1 << f);
    const float ay = ((hp + 0.5f) * 0.125f) * freq;
    const float ax = ((wp + 0.5f) * 0.125f) * freq;
    float v = (which == 0) ? __sinf(ay) : (which == 1) ? __cosf(ay)
            : (which == 2) ? __sinf(ax) : __cosf(ax);
    gfeat[64 + which * 4 + f] = v;
  }
  __syncthreads();
  if (t < 128) {
    float acc = Rb1[t];
    #pragma unroll 4
    for (int c = 0; c < 80; ++c) acc += gfeat[c] * Rw1[c * 128 + t];
    hid[t] = fmaxf(acc, 0.f);
  }
  __syncthreads();
  if (t < Ee) {
    float acc = Rb2[t];
    #pragma unroll 4
    for (int d = 0; d < 128; ++d) acc += hid[d] * Rw2[d * Ee + t];
    logit[t] = acc;
  }
  __syncthreads();
  if (t < Ee) {
    float m = logit[0];
    for (int j = 1; j < Ee; ++j) m = fmaxf(m, logit[j]);
    float den = 0.f;
    for (int j = 0; j < Ee; ++j) den += __expf(logit[j] - m);
    gates[bid * Ee + t] = __expf(logit[t] - m) / den;
  }
}

// ---------------- TDM: load A weight tile [128 rows x 32 K] bf16 -> LDS ----------------
// D# per CDNA5 ISA: group0 {count, lds_addr, global_addr, type=2},
// group1 {data_size=2B, tensor_dim0=Ktot, tensor_dim1=128, tile 32x128, dim0_stride=Ktot}.
__device__ __forceinline__ void tdm_load_A(const __bf16* gptr, unsigned ldsA, int Ktot) {
  unsigned long long ga = (unsigned long long)(const void*)gptr;
  u32x4 g0;
  g0[0] = 1u;                                   // count = 1 valid descriptor
  g0[1] = ldsA;                                 // LDS byte address
  g0[2] = (unsigned)ga;                         // global_addr[31:0]
  g0[3] = (unsigned)(ga >> 32) | (2u << 30);    // global_addr[56:32] | type=2
  i32x8 g1;
  g1[0] = 0x00010000;                           // data_size = 2 bytes (code 1)
  g1[1] = (Ktot & 0xFFFF) << 16;                // tensor_dim0[15:0]
  g1[2] = (128 << 16) | ((Ktot >> 16) & 0xFFFF);// tensor_dim1[15:0] | tensor_dim0[31:16]
  g1[3] = (32 << 16);                           // tile_dim0 = 32
  g1[4] = 128;                                  // tile_dim1 = 128
  g1[5] = Ktot;                                 // tensor_dim0_stride[31:0]
  g1[6] = 0;
  g1[7] = 0;
  i32x4 gz4 = {0, 0, 0, 0};
  i32x8 gz8 = {0, 0, 0, 0, 0, 0, 0, 0};
  __builtin_amdgcn_tensor_load_to_lds(g0, g1, gz4, gz4, gz8, 0);
}

// ---------------- branchless 8-element im2col gather into registers ----------------
// stab[k] = (ci*HW + kh*W + kw) | (r<<28); invalid halo lanes clamp offset to 0 and
// cndmask the loaded value to zero (clamped address stays inside the workspace).
template<int KS>
__device__ __forceinline__ void gather_b8(const __bf16* __restrict__ psrc,
                                          const unsigned* __restrict__ stab,
                                          int k0, unsigned vm, __bf16 (&tmp)[8]) {
  if (KS == 3) {
    u32x4 su0 = *((const u32x4*)&stab[k0]);
    u32x4 su1 = *((const u32x4*)&stab[k0 + 4]);
    #pragma unroll
    for (int j = 0; j < 8; ++j) {
      const unsigned u = (j < 4) ? su0[j & 3] : su1[j & 3];
      const unsigned r = u >> 28;
      const bool ok = ((vm >> r) & 1u) != 0u;
      const unsigned off = ok ? (u & 0x0FFFFFFFu) : 0u;
      const __bf16 v = psrc[off];
      tmp[j] = ok ? v : (__bf16)0.0f;
    }
  } else {
    #pragma unroll
    for (int j = 0; j < 8; ++j) tmp[j] = psrc[(size_t)(k0 + j) << 14];  // k*HW
  }
}

// ---------------- 4 back-to-back WMMAs from one A/B LDS tile ----------------
__device__ __forceinline__ void mma_step(const __bf16* __restrict__ Alds,
                                         const __bf16* __restrict__ Blds,
                                         int lane, int wid, f32x8 acc[4]) {
  FragBF a;
  const int mrow = wid * 16 + (lane & 15);
  const int kA0  = (lane < 16) ? 0 : 8;          // A frag: K{0..7,16..23} / K{8..15,24..31}
  a.q[0] = *((const u32x4*)&Alds[mrow * 32 + kA0]);
  a.q[1] = *((const u32x4*)&Alds[mrow * 32 + kA0 + 16]);
  const int kB0 = (lane >> 4) * 8;               // B frag: K{0..7,16..23} / K{8..15,24..31}
  FragBF bb[4];
  #pragma unroll
  for (int nh = 0; nh < 4; ++nh) {
    const int col = nh * 16 + (lane & 15);
    bb[nh].q[0] = *((const u32x4*)&Blds[col * 32 + kB0]);
    bb[nh].q[1] = *((const u32x4*)&Blds[col * 32 + kB0 + 16]);
  }
  #pragma unroll
  for (int nh = 0; nh < 4; ++nh)
    acc[nh] = __builtin_amdgcn_wmma_f32_16x16x32_bf16(
        false, a.v, false, bb[nh].v, (short)0, acc[nh], false, false);
}

// ---------------- implicit-GEMM inner loop (bf16 WMMA, fp32 acc) ----------------
// Tile: M=128 x N=64, K-step 32. Double-buffered LDS: TDM DMA of A tile s+1 and the
// global gather loads for B tile s+1 are issued before the WMMAs of tile s.
template<int KS>
__device__ __forceinline__ void gemm_accum(
    const __bf16* __restrict__ Wbf, int Ktot,
    const __bf16* __restrict__ src /* [Csrc][H][W] of this batch */,
    int h0, int w0,
    __bf16* __restrict__ Alds, __bf16* __restrict__ Blds,
    unsigned* __restrict__ stab, unsigned ldsA,
    f32x8 acc[4]) {
  constexpr int ATILE = 128 * 32;      // elements per A buffer
  constexpr int BTILE = 64 * 32;       // elements per B buffer
  const int t    = threadIdx.x;
  const int lane = t & 31;
  const int wid  = t >> 5;
  const int bN   = t >> 2;             // 0..63
  const int bK   = (t & 3) * 8;        // 0,8,16,24

  __syncthreads();                     // previous users of stab/Blds done
  if (KS == 3) {                       // decode table
    for (int k = t; k < Ktot; k += 256) {
      int ci = k / 9; int r = k - ci * 9; int kh = r / 3, kw = r - kh * 3;
      stab[k] = (unsigned)(ci * HW + kh * Ww + kw) | ((unsigned)r << 28);
    }
  }
  // per-thread source base + 9-bit halo validity mask (h0, bN fixed per thread)
  const __bf16* psrc;
  unsigned vm = 0;
  if (KS == 3) {
    psrc = src + (h0 - 1) * Ww + (w0 + bN - 1);
    #pragma unroll
    for (int r = 0; r < 9; ++r) {
      int kh = r / 3, kw = r - kh * 3;
      int ih = h0 + kh - 1, iw = w0 + bN + kw - 1;
      if ((unsigned)ih < (unsigned)Hh && (unsigned)iw < (unsigned)Ww) vm |= 1u << r;
    }
  } else {
    psrc = src + h0 * Ww + w0 + bN;    // 1x1: always in-bounds
  }
  __syncthreads();                     // publish stab

  // ---- prologue: stage tile 0 into buffer 0 ----
  if (wid == 0) tdm_load_A(Wbf, ldsA, Ktot);
  {
    alignas(16) __bf16 tmp[8];
    gather_b8<KS>(psrc, stab, bK, vm, tmp);
    *((u32x4*)&Blds[bN * 32 + bK]) = *((const u32x4*)&tmp[0]);
  }
  if (wid == 0) __builtin_amdgcn_s_wait_tensorcnt(0);
  __syncthreads();

  const int nsteps = Ktot / 32;
  for (int s = 0; s < nsteps; ++s) {
    const int cur = s & 1, nxt = cur ^ 1;
    const bool more = (s + 1) < nsteps;
    // prefetch tile s+1: TDM for A, global loads for B (held in registers)
    if (more && wid == 0)
      tdm_load_A(Wbf + (s + 1) * 32, ldsA + (unsigned)(nxt * ATILE * 2), Ktot);
    alignas(16) __bf16 tmp[8];
    if (more) gather_b8<KS>(psrc, stab, (s + 1) * 32 + bK, vm, tmp);
    // compute tile s while the DMA + gather loads are in flight
    mma_step(Alds + cur * ATILE, Blds + cur * BTILE, lane, wid, acc);
    // commit gathered B to the next buffer
    if (more) *((u32x4*)&Blds[nxt * BTILE + bN * 32 + bK]) = *((const u32x4*)&tmp[0]);
    if (wid == 0) __builtin_amdgcn_s_wait_tensorcnt(0);
    __syncthreads();
  }
}

// ---------------- expert stage 1: h1 = relu(conv1(x)+b1), bf16 ----------------
__global__ __launch_bounds__(256)
void conv1_kernel(const __bf16* __restrict__ Wbf1, const float* __restrict__ b1,
                  const __bf16* __restrict__ xbf, __bf16* __restrict__ h1, int e) {
  __shared__ alignas(16) __bf16 Alds[2 * 128 * 32];
  __shared__ alignas(16) __bf16 Blds[2 * 64 * 32];
  __shared__ alignas(16) unsigned stab[1152];
  const unsigned ldsA = (unsigned)(unsigned long long)(void*)Alds;
  const int n0 = blockIdx.x * 64;
  const int b = n0 >> 14, h0 = (n0 >> 7) & 127, w0 = n0 & 127;
  f32x8 acc[4] = {};
  gemm_accum<3>(Wbf1 + (size_t)e * COUT * (CIN * 9), CIN * 9,
                xbf + (size_t)b * CIN * HW, h0, w0, Alds, Blds, stab, ldsA, acc);
  const int t = threadIdx.x, lane = t & 31, wid = t >> 5;
  const float* bias = b1 + e * COUT;
  __bf16* outp = h1 + (size_t)b * COUT * HW;
  #pragma unroll
  for (int nh = 0; nh < 4; ++nh) {
    const int ncol = nh * 16 + (lane & 15);
    #pragma unroll
    for (int r = 0; r < 8; ++r) {
      const int m = wid * 16 + r + ((lane >> 4) << 3);
      float v = fmaxf(acc[nh][r] + bias[m], 0.f);
      outp[(size_t)m * HW + h0 * Ww + w0 + ncol] = (__bf16)v;
    }
  }
}

// ---------------- expert stage 2: y (+)= g_e * relu(conv2(h1)+b2 + res(x)+br) ----------------
__global__ __launch_bounds__(256)
void conv2res_kernel(const __bf16* __restrict__ Wbf2, const float* __restrict__ b2,
                     const __bf16* __restrict__ Wbfr, const float* __restrict__ br,
                     const __bf16* __restrict__ xbf, const __bf16* __restrict__ h1,
                     const float* __restrict__ gates, float* __restrict__ yacc,
                     int e, int first) {
  __shared__ alignas(16) __bf16 Alds[2 * 128 * 32];
  __shared__ alignas(16) __bf16 Blds[2 * 64 * 32];
  __shared__ alignas(16) unsigned stab[1152];
  const unsigned ldsA = (unsigned)(unsigned long long)(void*)Alds;
  const int n0 = blockIdx.x * 64;
  const int b = n0 >> 14, h0 = (n0 >> 7) & 127, w0 = n0 & 127;
  f32x8 acc[4] = {};
  gemm_accum<1>(Wbfr + (size_t)e * COUT * CIN, CIN,
                xbf + (size_t)b * CIN * HW, h0, w0, Alds, Blds, stab, ldsA, acc);
  gemm_accum<3>(Wbf2 + (size_t)e * COUT * (COUT * 9), COUT * 9,
                h1 + (size_t)b * COUT * HW, h0, w0, Alds, Blds, stab, ldsA, acc);
  const int t = threadIdx.x, lane = t & 31, wid = t >> 5;
  const float* bias2 = b2 + e * COUT;
  const float* biasr = br + e * COUT;
  float* outp = yacc + (size_t)b * COUT * HW;
  const int hp = h0 >> 4;
  #pragma unroll
  for (int nh = 0; nh < 4; ++nh) {
    const int ncol = nh * 16 + (lane & 15);
    const int wp = (w0 + ncol) >> 4;
    const float g = gates[(((b * HP) + hp) * WP + wp) * Ee + e];
    #pragma unroll
    for (int r = 0; r < 8; ++r) {
      const int m = wid * 16 + r + ((lane >> 4) << 3);
      const float v = fmaxf(acc[nh][r] + bias2[m] + biasr[m], 0.f) * g;
      const size_t idx = (size_t)m * HW + h0 * Ww + w0 + ncol;
      if (first) outp[idx] = v; else outp[idx] += v;
    }
  }
}

// ---------------- GroupNorm: per-(b,g) stats ----------------
__global__ __launch_bounds__(256)
void gn_stats_kernel(const float* __restrict__ yacc, float* __restrict__ stats) {
  __shared__ float s1[256], s2[256];
  const int grp = blockIdx.x;                    // group data is contiguous
  const f32x4* p4 = (const f32x4*)(yacc + (size_t)grp * (COUT / Gg) * HW);
  const int n4 = (COUT / Gg) * HW / 4;           // 65536
  float a = 0.f, q = 0.f;
  for (int i = threadIdx.x; i < n4; i += 256) {
    f32x4 v = p4[i];
    a += v[0] + v[1] + v[2] + v[3];
    q += v[0] * v[0] + v[1] * v[1] + v[2] * v[2] + v[3] * v[3];
  }
  s1[threadIdx.x] = a; s2[threadIdx.x] = q;
  __syncthreads();
  for (int off = 128; off > 0; off >>= 1) {
    if (threadIdx.x < off) {
      s1[threadIdx.x] += s1[threadIdx.x + off];
      s2[threadIdx.x] += s2[threadIdx.x + off];
    }
    __syncthreads();
  }
  if (threadIdx.x == 0) {
    const float inv = 1.f / (float)((COUT / Gg) * HW);
    const float mu  = s1[0] * inv;
    const float var = s2[0] * inv - mu * mu;
    stats[grp * 2]     = mu;
    stats[grp * 2 + 1] = rsqrtf(var + EPSF);
  }
}

// ---------------- GroupNorm apply + affine ----------------
__global__ __launch_bounds__(256)
void gn_apply_kernel(const float* __restrict__ yacc, const float* __restrict__ stats,
                     const float* __restrict__ gamma, const float* __restrict__ beta,
                     float* __restrict__ out) {
  const int i4 = blockIdx.x * 256 + threadIdx.x;
  const size_t base = (size_t)i4 * 4;
  const int co = (int)((base >> 14) & 127);
  const int b  = (int)(base >> 21);
  const int grp = b * Gg + (co >> 4);
  const float mu = stats[grp * 2], rs = stats[grp * 2 + 1];
  const float ga = gamma[co], be = beta[co];
  f32x4 v = ((const f32x4*)yacc)[i4];
  f32x4 o;
  #pragma unroll
  for (int j = 0; j < 4; ++j) o[j] = (v[j] - mu) * rs * ga + be;
  ((f32x4*)out)[i4] = o;
}

// ---------------- host launcher ----------------
extern "C" void kernel_launch(void* const* d_in, const int* in_sizes, int n_in,
                              void* d_out, int out_size, void* d_ws, size_t ws_size,
                              hipStream_t stream) {
  const float* x     = (const float*)d_in[0];
  const float* W1    = (const float*)d_in[1];
  const float* b1    = (const float*)d_in[2];
  const float* W2    = (const float*)d_in[3];
  const float* b2    = (const float*)d_in[4];
  const float* Wr    = (const float*)d_in[5];
  const float* br    = (const float*)d_in[6];
  const float* Rw1   = (const float*)d_in[7];
  const float* Rb1   = (const float*)d_in[8];
  const float* Rw2   = (const float*)d_in[9];
  const float* Rb2   = (const float*)d_in[10];
  const float* gamma = (const float*)d_in[11];
  const float* beta  = (const float*)d_in[12];
  float* out = (float*)d_out;

  // workspace layout
  char* ws = (char*)d_ws;
  size_t o = 0;
  float*  gates = (float*)(ws + o);  o += (16 << 10);
  float*  stats = (float*)(ws + o);  o += (16 << 10);
  __bf16* xbf   = (__bf16*)(ws + o); o += (size_t)Bb * CIN * HW * 2;    // 16 MB
  __bf16* h1    = (__bf16*)(ws + o); o += (size_t)Bb * COUT * HW * 2;   // 32 MB
  float*  yacc  = (float*)(ws + o);  o += (size_t)Bb * COUT * HW * 4;   // 64 MB
  __bf16* wbf1  = (__bf16*)(ws + o); o += (size_t)Ee * COUT * CIN * 9 * 2;
  __bf16* wbf2  = (__bf16*)(ws + o); o += (size_t)Ee * COUT * COUT * 9 * 2;
  __bf16* wbfr  = (__bf16*)(ws + o); o += (size_t)Ee * COUT * CIN * 2;

  const int nx4  = Bb * CIN * HW / 4;           // 2,097,152
  const int nw14 = Ee * COUT * CIN * 9 / 4;     // 147,456
  const int nw24 = Ee * COUT * COUT * 9 / 4;    // 294,912
  const int nwr4 = Ee * COUT * CIN / 4;         // 16,384
  cvt_bf16_kernel<<<nx4  / 256, 256, 0, stream>>>(x,  xbf,  nx4);
  cvt_bf16_kernel<<<nw14 / 256, 256, 0, stream>>>(W1, wbf1, nw14);
  cvt_bf16_kernel<<<nw24 / 256, 256, 0, stream>>>(W2, wbf2, nw24);
  cvt_bf16_kernel<<<nwr4 / 256, 256, 0, stream>>>(Wr, wbfr, nwr4);
  router_kernel<<<Bb * HP * WP, 256, 0, stream>>>(x, Rw1, Rb1, Rw2, Rb2, gates);

  const int nblk = (Bb * Hh * Ww) / 64;         // 2048
  for (int e = 0; e < Ee; ++e) {
    conv1_kernel   <<<nblk, 256, 0, stream>>>(wbf1, b1, xbf, h1, e);
    conv2res_kernel<<<nblk, 256, 0, stream>>>(wbf2, b2, wbfr, br, xbf, h1, gates, yacc, e, e == 0 ? 1 : 0);
  }
  gn_stats_kernel<<<Bb * Gg, 256, 0, stream>>>(yacc, stats);
  gn_apply_kernel<<<(Bb * COUT * HW / 4) / 256, 256, 0, stream>>>(yacc, stats, gamma, beta, out);
}
